// DecoderAblationAttModule_39402029974009
// MI455X (gfx1250) — matrix operand order repaired and checked by reading, not compiled, not run on hardware
//
#include <hip/hip_runtime.h>
#include <hip/hip_bf16.h>

// ---------------- problem constants (fixed shapes from reference) ----------
static constexpr int cB  = 256;    // batch
static constexpr int cN  = 36;     // regions
static constexpr int cFT = 2048;   // feature size
static constexpr int cTS = 20;     // caption length
static constexpr int cT  = 19;     // decode steps (T_SEQ - 1)
static constexpr int cA  = 512;    // attention size
static constexpr int cD  = 1024;   // decoder size
static constexpr int cE  = 512;    // embed size
static constexpr int cV  = 12000;  // vocab
static constexpr int cKG = cFT + cE + cD;  // 3584  gates GEMM K
static constexpr int cKP = cD + cFT;       // 3072  pred  GEMM K

typedef __attribute__((ext_vector_type(16))) __bf16 v16bf;
typedef __attribute__((ext_vector_type(8)))  float  v8f;

// ---------------- bf16 WMMA GEMM: C[M,N] = A[M,K] * W[N,K]^T (+bias1+bias2)
// Register-blocked: one wave -> 64(M) x 32(N) macro-tile = 4x2 WMMA tiles.
// 4 A-frags + 2 B-frags feed 8 WMMAs per K-step (0.75 fragment loads/WMMA).
// Block = 8 waves spanning 256 cols of N. Requires M % 64 == 0, N % 32 == 0,
// K % 32 == 0 (all shapes here satisfy this).
__global__ __launch_bounds__(256) void k_gemm_bf16(
    const __bf16* __restrict__ Abuf, const __bf16* __restrict__ Wbuf,
    float* __restrict__ Cbuf,
    const float* __restrict__ bias1, const float* __restrict__ bias2,
    int M, int N, int K, int lda, int ldw, int ldc)
{
    const int wave = threadIdx.x >> 5;
    const int lane = threadIdx.x & 31;
    const int m0 = blockIdx.y * 64;
    const int n0 = blockIdx.x * 256 + wave * 32;
    if (n0 >= N || m0 >= M) return;        // wave-uniform: EXEC stays full

    const int r    = lane & 15;
    const int half = lane >> 4;

    const __bf16* Arow[4];
    const __bf16* Wrow[2];
#pragma unroll
    for (int i = 0; i < 4; ++i)
        Arow[i] = Abuf + (size_t)(m0 + 16 * i + r) * lda;
#pragma unroll
    for (int j = 0; j < 2; ++j)
        Wrow[j] = Wbuf + (size_t)(n0 + 16 * j + r) * ldw;

    v8f acc[4][2] = {};
    for (int k0 = 0; k0 < K; k0 += 32) {
        v16bf a[4], b[2];
        // A (16x32 bf16): lanes 0-15 K=0..7 / 16..23 ; lanes 16-31 +8
#pragma unroll
        for (int i = 0; i < 4; ++i) {
#pragma unroll
            for (int j = 0; j < 4; ++j) {
                int ka = k0 + 2 * j + 8 * half;
                a[i][2 * j]     = Arow[i][ka];
                a[i][2 * j + 1] = Arow[i][ka + 1];
                int ka2 = k0 + 16 + 2 * j + 8 * half;
                a[i][8 + 2 * j]     = Arow[i][ka2];
                a[i][8 + 2 * j + 1] = Arow[i][ka2 + 1];
            }
        }
        // B (32x16 bf16): lanes 0-15 K=0..7 / 16..23 ; lanes 16-31 +8
#pragma unroll
        for (int jj = 0; jj < 2; ++jj) {
#pragma unroll
            for (int j = 0; j < 4; ++j) {
                int kb = k0 + 2 * j + 8 * half;
                b[jj][2 * j]     = Wrow[jj][kb];
                b[jj][2 * j + 1] = Wrow[jj][kb + 1];
                int kb2 = k0 + 16 + 2 * j + 8 * half;
                b[jj][8 + 2 * j]     = Wrow[jj][kb2];
                b[jj][8 + 2 * j + 1] = Wrow[jj][kb2 + 1];
            }
        }
#pragma unroll
        for (int i = 0; i < 4; ++i)
#pragma unroll
            for (int jj = 0; jj < 2; ++jj)
                acc[i][jj] = __builtin_amdgcn_wmma_f32_16x16x32_bf16(
                    false, a[i], false, b[jj], (short)0, acc[i][jj],
                    false, false);
    }

    // epilogue: fused biases
#pragma unroll
    for (int jj = 0; jj < 2; ++jj) {
        int n = n0 + 16 * jj + r;
        float badd = 0.f;
        if (bias1) badd += bias1[n];
        if (bias2) badd += bias2[n];
#pragma unroll
        for (int i = 0; i < 4; ++i) {
#pragma unroll
            for (int j = 0; j < 8; ++j) {
                int m = m0 + 16 * i + j + 8 * half;  // C: VGPR j -> row j (+8 hi half)
                Cbuf[(size_t)m * ldc + n] = acc[i][jj][j] + badd;
            }
        }
    }
}

// ---------------- setup kernels --------------------------------------------
// stable descending sort of sizes -> order, dec_len
__global__ void k_sort(const int* __restrict__ sizes, int* __restrict__ order,
                       int* __restrict__ dec_len)
{
    int i = threadIdx.x;
    if (i >= cB) return;
    int si = sizes[i];
    int pos = 0;
    for (int j = 0; j < cB; ++j) {
        int sj = sizes[j];
        if (sj > si || (sj == si && j < i)) pos++;
    }
    order[pos]   = i;
    dec_len[pos] = si - 1;
}

__global__ void k_gather_feats_bf16(const float* __restrict__ feats,
                                    const int* __restrict__ order,
                                    __bf16* __restrict__ out)
{
    long long idx = (long long)blockIdx.x * blockDim.x + threadIdx.x;
    long long total = (long long)cB * cN * cFT;
    if (idx >= total) return;
    int b = (int)(idx / ((long long)cN * cFT));
    long long r = idx % ((long long)cN * cFT);
    out[idx] = (__bf16)feats[(size_t)order[b] * cN * cFT + r];
}

__global__ void k_f32_to_bf16(const float* __restrict__ src,
                              __bf16* __restrict__ dst, long long n)
{
    long long i = (long long)blockIdx.x * blockDim.x + threadIdx.x;
    if (i < n) dst[i] = (__bf16)src[i];
}

// dst[rows, k1+k2] = [src1 | src2] rows-major, cast to bf16
__global__ void k_cat2_bf16(const float* __restrict__ s1,
                            const float* __restrict__ s2,
                            __bf16* __restrict__ dst,
                            int rows, int k1, int k2)
{
    long long idx = (long long)blockIdx.x * blockDim.x + threadIdx.x;
    long long total = (long long)rows * (k1 + k2);
    if (idx >= total) return;
    int kk = k1 + k2;
    int r = (int)(idx / kk);
    int c = (int)(idx % kk);
    float v = (c < k1) ? s1[(size_t)r * k1 + c] : s2[(size_t)r * k2 + (c - k1)];
    dst[idx] = (__bf16)v;
}

__global__ void k_zero_f32(float* __restrict__ p, long long n)
{
    long long i = (long long)blockIdx.x * blockDim.x + threadIdx.x;
    if (i < n) p[i] = 0.f;
}

// ---------------- per-step kernels -----------------------------------------
// scores[b,n] = sum_a relu(h_att[b,a] + img_att[b,n,a]) * Wa[a] + ba
__global__ void k_att_scores(const float* __restrict__ h_att,
                             const float* __restrict__ img_att,
                             const float* __restrict__ Wa,
                             const float* __restrict__ ba,
                             float* __restrict__ scores)
{
    int n = blockIdx.x, b = blockIdx.y, tid = threadIdx.x;
    const float* ia = img_att + ((size_t)b * cN + n) * cA;
    const float* ha = h_att + (size_t)b * cA;
    float s = 0.f;
    for (int a = tid; a < cA; a += 64) {
        float v = ha[a] + ia[a];
        v = v > 0.f ? v : 0.f;
        s += v * Wa[a];
    }
    __shared__ float red[64];
    red[tid] = s;
    __syncthreads();
    for (int off = 32; off > 0; off >>= 1) {
        if (tid < off) red[tid] += red[tid + off];
        __syncthreads();
    }
    if (tid == 0) scores[b * cN + n] = red[0] + ba[0];
}

__global__ void k_softmax(const float* __restrict__ scores,
                          float* __restrict__ alpha)
{
    int b = blockIdx.x * blockDim.x + threadIdx.x;
    if (b >= cB) return;
    const float* s = scores + (size_t)b * cN;
    float m = -1e30f;
    for (int n = 0; n < cN; ++n) m = fmaxf(m, s[n]);
    float e[cN];
    float sum = 0.f;
    for (int n = 0; n < cN; ++n) { e[n] = __expf(s[n] - m); sum += e[n]; }
    float inv = 1.f / sum;
    for (int n = 0; n < cN; ++n) alpha[(size_t)b * cN + n] = e[n] * inv;
}

// aw[b,f] = sum_n feats_bf[b,n,f] * alpha[b,n]
__global__ void k_aw(const __bf16* __restrict__ feats_bf,
                     const float* __restrict__ alpha,
                     float* __restrict__ aw)
{
    int b = blockIdx.y;
    int f = blockIdx.x * 256 + threadIdx.x;
    __shared__ float al[cN];
    if (threadIdx.x < cN) al[threadIdx.x] = alpha[(size_t)b * cN + threadIdx.x];
    __syncthreads();
    float s = 0.f;
#pragma unroll 4
    for (int n = 0; n < cN; ++n)
        s += (float)feats_bf[((size_t)b * cN + n) * cFT + f] * al[n];
    aw[(size_t)b * cFT + f] = s;
}

// xg[b, 0:2048]=aw ; [2048:2560]=emb[seq[order[b],t]] ; [2560:3584]=hidden
__global__ void k_build_xg(const float* __restrict__ aw,
                           const float* __restrict__ emb,
                           const int* __restrict__ sequences,
                           const int* __restrict__ order,
                           const float* __restrict__ hidden,
                           __bf16* __restrict__ xg, int t)
{
    long long idx = (long long)blockIdx.x * blockDim.x + threadIdx.x;
    long long total = (long long)cB * cKG;
    if (idx >= total) return;
    int b = (int)(idx / cKG);
    int c = (int)(idx % cKG);
    float v;
    if (c < cFT) {
        v = aw[(size_t)b * cFT + c];
    } else if (c < cFT + cE) {
        int tok = sequences[(size_t)order[b] * cTS + t];
        v = emb[(size_t)tok * cE + (c - cFT)];
    } else {
        v = hidden[(size_t)b * cD + (c - cFT - cE)];
    }
    xg[idx] = (__bf16)v;
}

__global__ void k_lstm(const float* __restrict__ gates,
                       const int* __restrict__ dec_len, int t,
                       float* __restrict__ hidden, float* __restrict__ cell,
                       float* __restrict__ h_new)
{
    long long idx = (long long)blockIdx.x * blockDim.x + threadIdx.x;
    if (idx >= (long long)cB * cD) return;
    int b = (int)(idx / cD);
    int d = (int)(idx % cD);
    const float* g = gates + (size_t)b * 4 * cD;
    float gi = g[d], gf = g[cD + d], gg = g[2 * cD + d], go = g[3 * cD + d];
    float si = 1.f / (1.f + __expf(-gi));
    float sf = 1.f / (1.f + __expf(-gf));
    float so = 1.f / (1.f + __expf(-go));
    float cn = sf * cell[idx] + si * tanhf(gg);
    float hn = so * tanhf(cn);
    h_new[idx] = hn;
    if (dec_len[b] > t) { hidden[idx] = hn; cell[idx] = cn; }
}

// xp[b, 0:1024]=h_new ; [1024:3072]=aw
__global__ void k_build_xp(const float* __restrict__ h_new,
                           const float* __restrict__ aw,
                           __bf16* __restrict__ xp)
{
    long long idx = (long long)blockIdx.x * blockDim.x + threadIdx.x;
    long long total = (long long)cB * cKP;
    if (idx >= total) return;
    int b = (int)(idx / cKP);
    int c = (int)(idx % cKP);
    float v = (c < cD) ? h_new[(size_t)b * cD + c]
                       : aw[(size_t)b * cFT + (c - cD)];
    xp[idx] = (__bf16)v;
}

// d_out[b, t, v] = active ? predbuf[b, v] : 0   (biases fused in GEMM)
__global__ void k_pred_out(const float* __restrict__ predbuf,
                           const int* __restrict__ dec_len, int t,
                           float* __restrict__ out)
{
    long long idx = (long long)blockIdx.x * blockDim.x + threadIdx.x;
    if (idx >= (long long)cB * cV) return;
    int b = (int)(idx / cV);
    int v = (int)(idx % cV);
    float val = (dec_len[b] > t) ? predbuf[idx] : 0.f;
    out[((size_t)b * cT + t) * cV + v] = val;
}

// ---------------- host launcher --------------------------------------------
extern "C" void kernel_launch(void* const* d_in, const int* in_sizes, int n_in,
                              void* d_out, int out_size, void* d_ws, size_t ws_size,
                              hipStream_t stream)
{
    const float* feats     = (const float*)d_in[0];
    const int*   sequences = (const int*)d_in[1];
    const int*   sizes     = (const int*)d_in[2];
    const float* emb       = (const float*)d_in[3];
    const float* Wf        = (const float*)d_in[4];
    const float* bf        = (const float*)d_in[5];
    const float* Wd        = (const float*)d_in[6];
    const float* bd        = (const float*)d_in[7];
    const float* Wa        = (const float*)d_in[8];
    const float* ba        = (const float*)d_in[9];
    const float* Wih       = (const float*)d_in[10];
    const float* Whh       = (const float*)d_in[11];
    const float* bih       = (const float*)d_in[12];
    const float* bhh       = (const float*)d_in[13];
    const float* Wh        = (const float*)d_in[14];
    const float* bh        = (const float*)d_in[15];
    const float* Wi        = (const float*)d_in[16];
    const float* bi        = (const float*)d_in[17];
    float* out = (float*)d_out;

    // ---- carve workspace
    char* p = (char*)d_ws;
    auto alloc = [&](size_t bytes) -> void* {
        void* r = (void*)p;
        p += (bytes + 255) & ~(size_t)255;
        return r;
    };
    int*    order     = (int*)alloc(cB * sizeof(int));
    int*    dec_len   = (int*)alloc(cB * sizeof(int));
    __bf16* feats_bf  = (__bf16*)alloc((size_t)cB * cN * cFT * 2);
    __bf16* Wf_bf     = (__bf16*)alloc((size_t)cA * cFT * 2);
    __bf16* Wd_bf     = (__bf16*)alloc((size_t)cA * cD * 2);
    __bf16* Wg_cat    = (__bf16*)alloc((size_t)4 * cD * cKG * 2);
    __bf16* Wp_cat    = (__bf16*)alloc((size_t)cV * cKP * 2);
    float*  img_att   = (float*)alloc((size_t)cB * cN * cA * 4);
    float*  hidden    = (float*)alloc((size_t)cB * cD * 4);
    float*  cell      = (float*)alloc((size_t)cB * cD * 4);
    float*  h_new     = (float*)alloc((size_t)cB * cD * 4);
    __bf16* hidden_bf = (__bf16*)alloc((size_t)cB * cD * 2);
    float*  h_att     = (float*)alloc((size_t)cB * cA * 4);
    float*  scores    = (float*)alloc((size_t)cB * cN * 4);
    float*  alpha     = (float*)alloc((size_t)cB * cN * 4);
    float*  aw        = (float*)alloc((size_t)cB * cFT * 4);
    __bf16* xg        = (__bf16*)alloc((size_t)cB * cKG * 2);
    float*  gates     = (float*)alloc((size_t)cB * 4 * cD * 4);
    __bf16* xp        = (__bf16*)alloc((size_t)cB * cKP * 2);
    float*  predbuf   = (float*)alloc((size_t)cB * cV * 4);
    (void)ws_size; (void)n_in; (void)in_sizes; (void)out_size;

    auto blocks = [](long long n) { return (unsigned)((n + 255) / 256); };
    auto gemm = [&](const __bf16* Ab, int lda, const __bf16* Wb, int ldw,
                    float* Cb, int ldc, const float* b1, const float* b2,
                    int M, int N, int K) {
        dim3 grid((N + 255) / 256, (M + 63) / 64);
        k_gemm_bf16<<<grid, 256, 0, stream>>>(Ab, Wb, Cb, b1, b2,
                                              M, N, K, lda, ldw, ldc);
    };

    // ---- one-time setup
    k_sort<<<1, 256, 0, stream>>>(sizes, order, dec_len);
    k_gather_feats_bf16<<<blocks((long long)cB * cN * cFT), 256, 0, stream>>>(
        feats, order, feats_bf);
    k_f32_to_bf16<<<blocks((long long)cA * cFT), 256, 0, stream>>>(
        Wf, Wf_bf, (long long)cA * cFT);
    k_f32_to_bf16<<<blocks((long long)cA * cD), 256, 0, stream>>>(
        Wd, Wd_bf, (long long)cA * cD);
    k_cat2_bf16<<<blocks((long long)4 * cD * cKG), 256, 0, stream>>>(
        Wih, Whh, Wg_cat, 4 * cD, cFT + cE, cD);
    k_cat2_bf16<<<blocks((long long)cV * cKP), 256, 0, stream>>>(
        Wh, Wi, Wp_cat, cV, cD, cFT);
    k_zero_f32<<<blocks((long long)cB * cD), 256, 0, stream>>>(
        hidden, (long long)cB * cD);
    k_zero_f32<<<blocks((long long)cB * cD), 256, 0, stream>>>(
        cell, (long long)cB * cD);

    // img_att[b*n, a] = feats_s @ Wf^T + bf   (timestep-invariant, hoisted)
    gemm(feats_bf, cFT, Wf_bf, cFT, img_att, cA, bf, nullptr,
         cB * cN, cA, cFT);

    // ---- decode loop
    for (int t = 0; t < cT; ++t) {
        k_f32_to_bf16<<<blocks((long long)cB * cD), 256, 0, stream>>>(
            hidden, hidden_bf, (long long)cB * cD);
        // h_att = hidden @ Wd^T + bd
        gemm(hidden_bf, cD, Wd_bf, cD, h_att, cA, bd, nullptr, cB, cA, cD);
        // attention
        k_att_scores<<<dim3(cN, cB), 64, 0, stream>>>(h_att, img_att, Wa, ba,
                                                      scores);
        k_softmax<<<1, 256, 0, stream>>>(scores, alpha);
        k_aw<<<dim3(cFT / 256, cB), 256, 0, stream>>>(feats_bf, alpha, aw);
        // gates = [aw; emb_t; hidden] @ [Wih | Whh]^T + bih + bhh
        k_build_xg<<<blocks((long long)cB * cKG), 256, 0, stream>>>(
            aw, emb, sequences, order, hidden, xg, t);
        gemm(xg, cKG, Wg_cat, cKG, gates, 4 * cD, bih, bhh, cB, 4 * cD, cKG);
        // LSTM pointwise + masked state update
        k_lstm<<<blocks((long long)cB * cD), 256, 0, stream>>>(
            gates, dec_len, t, hidden, cell, h_new);
        // pred = [h_new; aw] @ [Wh | Wi]^T + bh + bi
        k_build_xp<<<blocks((long long)cB * cKP), 256, 0, stream>>>(h_new, aw, xp);
        gemm(xp, cKP, Wp_cat, cKP, predbuf, cV, bh, bi, cB, cV, cKP);
        k_pred_out<<<blocks((long long)cB * cV), 256, 0, stream>>>(
            predbuf, dec_len, t, out);
    }
}
